// InteractionNetwork_12884901888318
// MI455X (gfx1250) — compile-verified, matrix operand order checked by reading, not run on hardware
//
#include <hip/hip_runtime.h>

#define NN 50000
#define NE 500000
#define DD 128
#define HH 256
#define EPSV 1e-5f

typedef __attribute__((ext_vector_type(16))) __bf16 v16bf;
typedef __attribute__((ext_vector_type(8)))  float  v8f;
typedef unsigned short u16;
typedef unsigned int   u32;

__device__ __forceinline__ u16 f2bf(float f) {
  union { float f; u32 u; } v; v.f = f;
  u32 u = v.u;
  u32 r = (u + 0x7FFFu + ((u >> 16) & 1u)) >> 16;   // round-to-nearest-even
  return (u16)r;
}

union AFrag { u32 u[8]; v16bf v; };

// ---------------- weight repack: fp32 row-major [K,N] -> bf16 B-fragment order ----
// dst[((nt*ktiles + kt)*32 + lane)*16 + i] = W[k, n]
//   n = nt*16 + (lane&15)
//   k = kt*32 + (i<8 ? i : i+8) + 8*(lane>=16)
__global__ void wfrag_kernel(const float* __restrict__ W, u16* __restrict__ dst,
                             int K, int N, int total) {
  int idx = blockIdx.x * blockDim.x + threadIdx.x;
  if (idx >= total) return;
  int ktiles = K >> 5;
  int i    = idx & 15;
  int l    = (idx >> 4) & 31;
  int frag = idx >> 9;
  int kt   = frag % ktiles;
  int nt   = frag / ktiles;
  int hlf  = l >> 4;
  int n    = nt * 16 + (l & 15);
  int k    = kt * 32 + (i < 8 ? i : i + 8) + hlf * 8;
  dst[idx] = f2bf(W[(long)k * N + n]);
}

__global__ void zero_kernel(float* __restrict__ p, long total) {
  long idx = (long)blockIdx.x * blockDim.x + threadIdx.x;
  if (idx < total) p[idx] = 0.f;
}

// ---------------- edge MLP + scatter-add : 32 edges/block, 8 waves ----------------
__global__ __launch_bounds__(256)
void edge_kernel(const float* __restrict__ nodes, const float* __restrict__ edges,
                 const int* __restrict__ senders, const int* __restrict__ receivers,
                 const u16* __restrict__ W1, const float* __restrict__ b1,
                 const float* __restrict__ g, const float* __restrict__ bt,
                 const u16* __restrict__ W2, const float* __restrict__ b2,
                 float* __restrict__ edges_out, float* __restrict__ agg) {
  __shared__ int   s_snd[32], s_rcv[32];
  __shared__ u32   s_x[32 * 194];      // packed bf16 x-tile: 384 cols = 192 dw (+2 pad)
  __shared__ float s_h[32 * 260];      // fp32 hidden 32x256 (+4 pad)
  __shared__ float s_rs[256], s_rq[256];
  __shared__ float s_mu[32], s_ri[32];
  __shared__ u32   s_hb[32 * 130];     // bf16 hidden for GEMM2 A

  const int tid  = threadIdx.x;
  const int wave = tid >> 5;           // 0..7
  const int lane = tid & 31;
  const int row  = lane & 15;
  const int hlf  = lane >> 4;
  const long e0  = (long)blockIdx.x * 32;

  if (tid < 32) { s_snd[tid] = senders[e0 + tid]; s_rcv[tid] = receivers[e0 + tid]; }
  __syncthreads();

  // gather x = [edge | n_send | n_recv] -> packed bf16 LDS
  for (int d = tid; d < 32 * 192; d += 256) {
    int r = d / 192, c = d % 192;
    int k = c << 1;
    const float* src; int off;
    if (k < DD)            { src = edges + (e0 + r) * DD;            off = k;          }
    else if (k < 2 * DD)   { src = nodes + (long)s_snd[r] * DD;      off = k - DD;     }
    else                   { src = nodes + (long)s_rcv[r] * DD;      off = k - 2 * DD; }
    s_x[r * 194 + c] = (u32)f2bf(src[off]) | ((u32)f2bf(src[off + 1]) << 16);
  }
  __syncthreads();

  // GEMM1: [32x384] @ [384x256]; wave owns 32 cols x 32 rows (2 ntiles x 2 mtiles)
  v8f acc[2][2];                        // [nt][mt]
  #pragma unroll
  for (int nt = 0; nt < 2; ++nt)
    #pragma unroll
    for (int mt = 0; mt < 2; ++mt)
      #pragma unroll
      for (int v = 0; v < 8; ++v) acc[nt][mt][v] = 0.f;

  #pragma unroll 2
  for (int kt = 0; kt < 12; ++kt) {
    AFrag a[2];
    #pragma unroll
    for (int mt = 0; mt < 2; ++mt) {
      int base = (mt * 16 + row) * 194 + kt * 16 + hlf * 4;
      #pragma unroll
      for (int j = 0; j < 8; ++j) a[mt].u[j] = s_x[base + j + ((j >= 4) ? 4 : 0)];
    }
    #pragma unroll
    for (int nt = 0; nt < 2; ++nt) {
      int ntg = wave * 2 + nt;
      v16bf b = *(const v16bf*)(W1 + (((long)(ntg * 12 + kt) * 32 + lane) << 4));
      #pragma unroll
      for (int mt = 0; mt < 2; ++mt)
        acc[nt][mt] = __builtin_amdgcn_wmma_f32_16x16x32_bf16(false, a[mt].v, false, b,
                                                              (short)0, acc[nt][mt],
                                                              false, false);
    }
  }

  // h + b1 -> LDS fp32
  #pragma unroll
  for (int nt = 0; nt < 2; ++nt) {
    int n = (wave * 2 + nt) * 16 + row;
    float bb = b1[n];
    #pragma unroll
    for (int mt = 0; mt < 2; ++mt)
      #pragma unroll
      for (int v = 0; v < 8; ++v)
        s_h[(mt * 16 + v + hlf * 8) * 260 + n] = acc[nt][mt][v] + bb;
  }
  __syncthreads();

  // LayerNorm stats over 256 cols per row (32 rows, 8 partials each)
  {
    int r = tid >> 3, c0 = (tid & 7) * 32;
    float s = 0.f, q = 0.f;
    #pragma unroll
    for (int c = 0; c < 32; ++c) { float x = s_h[r * 260 + c0 + c]; s += x; q += x * x; }
    s_rs[tid] = s; s_rq[tid] = q;
  }
  __syncthreads();
  if (tid < 32) {
    float s = 0.f, q = 0.f;
    #pragma unroll
    for (int j = 0; j < 8; ++j) { s += s_rs[tid * 8 + j]; q += s_rq[tid * 8 + j]; }
    float mu = s * (1.f / HH);
    float var = q * (1.f / HH) - mu * mu;
    s_mu[tid] = mu; s_ri[tid] = rsqrtf(var + EPSV);
  }
  __syncthreads();

  // LN apply + ReLU -> bf16 LDS
  {
    int r = tid >> 3, c0 = (tid & 7) * 32;
    float mu = s_mu[r], ri = s_ri[r];
    #pragma unroll
    for (int c = c0; c < c0 + 32; c += 2) {
      float x0 = fmaxf((s_h[r * 260 + c]     - mu) * ri * g[c]     + bt[c],     0.f);
      float x1 = fmaxf((s_h[r * 260 + c + 1] - mu) * ri * g[c + 1] + bt[c + 1], 0.f);
      s_hb[r * 130 + (c >> 1)] = (u32)f2bf(x0) | ((u32)f2bf(x1) << 16);
    }
  }
  __syncthreads();

  // GEMM2: [32x256] @ [256x128]; wave owns 16 cols x 32 rows
  v8f acc2[2];                          // [mt]
  #pragma unroll
  for (int mt = 0; mt < 2; ++mt)
    #pragma unroll
    for (int v = 0; v < 8; ++v) acc2[mt][v] = 0.f;

  #pragma unroll
  for (int kt = 0; kt < 8; ++kt) {
    v16bf b = *(const v16bf*)(W2 + (((long)(wave * 8 + kt) * 32 + lane) << 4));
    #pragma unroll
    for (int mt = 0; mt < 2; ++mt) {
      AFrag a;
      int base = (mt * 16 + row) * 130 + kt * 16 + hlf * 4;
      #pragma unroll
      for (int j = 0; j < 8; ++j) a.u[j] = s_hb[base + j + ((j >= 4) ? 4 : 0)];
      acc2[mt] = __builtin_amdgcn_wmma_f32_16x16x32_bf16(false, a.v, false, b,
                                                         (short)0, acc2[mt], false, false);
    }
  }

  // epilogue: residual store + scatter-add messages
  {
    int n = wave * 16 + row;
    float bb = b2[n];
    #pragma unroll
    for (int mt = 0; mt < 2; ++mt)
      #pragma unroll
      for (int v = 0; v < 8; ++v) {
        int M = mt * 16 + v + hlf * 8;
        long e = e0 + M;
        float val = acc2[mt][v] + bb;
        edges_out[e * DD + n] = val + edges[e * DD + n];
        atomicAdd(&agg[(long)s_rcv[M] * DD + n], val);
      }
  }
}

// ---------------- node MLP : 32 nodes/block, 8 waves ----------------
__global__ __launch_bounds__(256)
void node_kernel(const float* __restrict__ nodes, const float* __restrict__ agg,
                 const u16* __restrict__ W1, const float* __restrict__ b1,
                 const float* __restrict__ g, const float* __restrict__ bt,
                 const u16* __restrict__ W2, const float* __restrict__ b2,
                 float* __restrict__ nodes_out) {
  __shared__ u32   s_x[32 * 130];      // packed bf16 x-tile: 256 cols = 128 dw (+2 pad)
  __shared__ float s_h[32 * 260];
  __shared__ float s_rs[256], s_rq[256];
  __shared__ float s_mu[32], s_ri[32];
  __shared__ u32   s_hb[32 * 130];

  const int tid  = threadIdx.x;
  const int wave = tid >> 5;
  const int lane = tid & 31;
  const int row  = lane & 15;
  const int hlf  = lane >> 4;
  const long n0  = (long)blockIdx.x * 32;
  const int nval = (int)((NN - n0) < 32 ? (NN - n0) : 32);

  // x = [node | agg] -> packed bf16 LDS (zero-fill invalid rows)
  for (int d = tid; d < 32 * 128; d += 256) {
    int r = d >> 7, c = d & 127;
    u32 pk = 0u;
    if (r < nval) {
      int k = c << 1;
      const float* src; int off;
      if (k < DD) { src = nodes + (n0 + r) * DD; off = k; }
      else        { src = agg   + (n0 + r) * DD; off = k - DD; }
      pk = (u32)f2bf(src[off]) | ((u32)f2bf(src[off + 1]) << 16);
    }
    s_x[r * 130 + c] = pk;
  }
  __syncthreads();

  // GEMM1: [32x256] @ [256x256]; wave owns 32 cols x 32 rows
  v8f acc[2][2];
  #pragma unroll
  for (int nt = 0; nt < 2; ++nt)
    #pragma unroll
    for (int mt = 0; mt < 2; ++mt)
      #pragma unroll
      for (int v = 0; v < 8; ++v) acc[nt][mt][v] = 0.f;

  #pragma unroll 2
  for (int kt = 0; kt < 8; ++kt) {
    AFrag a[2];
    #pragma unroll
    for (int mt = 0; mt < 2; ++mt) {
      int base = (mt * 16 + row) * 130 + kt * 16 + hlf * 4;
      #pragma unroll
      for (int j = 0; j < 8; ++j) a[mt].u[j] = s_x[base + j + ((j >= 4) ? 4 : 0)];
    }
    #pragma unroll
    for (int nt = 0; nt < 2; ++nt) {
      int ntg = wave * 2 + nt;
      v16bf b = *(const v16bf*)(W1 + (((long)(ntg * 8 + kt) * 32 + lane) << 4));
      #pragma unroll
      for (int mt = 0; mt < 2; ++mt)
        acc[nt][mt] = __builtin_amdgcn_wmma_f32_16x16x32_bf16(false, a[mt].v, false, b,
                                                              (short)0, acc[nt][mt],
                                                              false, false);
    }
  }

  #pragma unroll
  for (int nt = 0; nt < 2; ++nt) {
    int n = (wave * 2 + nt) * 16 + row;
    float bb = b1[n];
    #pragma unroll
    for (int mt = 0; mt < 2; ++mt)
      #pragma unroll
      for (int v = 0; v < 8; ++v)
        s_h[(mt * 16 + v + hlf * 8) * 260 + n] = acc[nt][mt][v] + bb;
  }
  __syncthreads();

  {
    int r = tid >> 3, c0 = (tid & 7) * 32;
    float s = 0.f, q = 0.f;
    #pragma unroll
    for (int c = 0; c < 32; ++c) { float x = s_h[r * 260 + c0 + c]; s += x; q += x * x; }
    s_rs[tid] = s; s_rq[tid] = q;
  }
  __syncthreads();
  if (tid < 32) {
    float s = 0.f, q = 0.f;
    #pragma unroll
    for (int j = 0; j < 8; ++j) { s += s_rs[tid * 8 + j]; q += s_rq[tid * 8 + j]; }
    float mu = s * (1.f / HH);
    float var = q * (1.f / HH) - mu * mu;
    s_mu[tid] = mu; s_ri[tid] = rsqrtf(var + EPSV);
  }
  __syncthreads();

  {
    int r = tid >> 3, c0 = (tid & 7) * 32;
    float mu = s_mu[r], ri = s_ri[r];
    #pragma unroll
    for (int c = c0; c < c0 + 32; c += 2) {
      float x0 = fmaxf((s_h[r * 260 + c]     - mu) * ri * g[c]     + bt[c],     0.f);
      float x1 = fmaxf((s_h[r * 260 + c + 1] - mu) * ri * g[c + 1] + bt[c + 1], 0.f);
      s_hb[r * 130 + (c >> 1)] = (u32)f2bf(x0) | ((u32)f2bf(x1) << 16);
    }
  }
  __syncthreads();

  // GEMM2: [32x256] @ [256x128]; wave owns 16 cols x 32 rows
  v8f acc2[2];
  #pragma unroll
  for (int mt = 0; mt < 2; ++mt)
    #pragma unroll
    for (int v = 0; v < 8; ++v) acc2[mt][v] = 0.f;

  #pragma unroll
  for (int kt = 0; kt < 8; ++kt) {
    v16bf b = *(const v16bf*)(W2 + (((long)(wave * 8 + kt) * 32 + lane) << 4));
    #pragma unroll
    for (int mt = 0; mt < 2; ++mt) {
      AFrag a;
      int base = (mt * 16 + row) * 130 + kt * 16 + hlf * 4;
      #pragma unroll
      for (int j = 0; j < 8; ++j) a.u[j] = s_hb[base + j + ((j >= 4) ? 4 : 0)];
      acc2[mt] = __builtin_amdgcn_wmma_f32_16x16x32_bf16(false, a.v, false, b,
                                                         (short)0, acc2[mt], false, false);
    }
  }

  {
    int n = wave * 16 + row;
    float bb = b2[n];
    #pragma unroll
    for (int mt = 0; mt < 2; ++mt)
      #pragma unroll
      for (int v = 0; v < 8; ++v) {
        int M = mt * 16 + v + hlf * 8;
        long r = n0 + M;
        if (M < nval)
          nodes_out[r * DD + n] = acc2[mt][v] + bb + nodes[r * DD + n];
      }
  }
}

extern "C" void kernel_launch(void* const* d_in, const int* in_sizes, int n_in,
                              void* d_out, int out_size, void* d_ws, size_t ws_size,
                              hipStream_t stream) {
  const float* nodes     = (const float*)d_in[0];
  const float* edges     = (const float*)d_in[1];
  const int*   senders   = (const int*)d_in[2];
  const int*   receivers = (const int*)d_in[3];
  const float* eW1 = (const float*)d_in[4];
  const float* eb1 = (const float*)d_in[5];
  const float* eg  = (const float*)d_in[6];
  const float* ebt = (const float*)d_in[7];
  const float* eW2 = (const float*)d_in[8];
  const float* eb2 = (const float*)d_in[9];
  const float* nW1 = (const float*)d_in[10];
  const float* nb1 = (const float*)d_in[11];
  const float* ng  = (const float*)d_in[12];
  const float* nbt = (const float*)d_in[13];
  const float* nW2 = (const float*)d_in[14];
  const float* nb2 = (const float*)d_in[15];

  float* nodes_out = (float*)d_out;
  float* edges_out = nodes_out + (long)NN * DD;

  float* agg  = (float*)d_ws;
  u16* eW1b = (u16*)(agg + (long)NN * DD);
  u16* eW2b = eW1b + 3 * DD * HH;
  u16* nW1b = eW2b + HH * DD;
  u16* nW2b = nW1b + 2 * DD * HH;

  zero_kernel<<<(NN * DD + 255) / 256, 256, 0, stream>>>(agg, (long)NN * DD);
  wfrag_kernel<<<(3 * DD * HH + 255) / 256, 256, 0, stream>>>(eW1, eW1b, 3 * DD, HH, 3 * DD * HH);
  wfrag_kernel<<<(HH * DD + 255) / 256, 256, 0, stream>>>(eW2, eW2b, HH, DD, HH * DD);
  wfrag_kernel<<<(2 * DD * HH + 255) / 256, 256, 0, stream>>>(nW1, nW1b, 2 * DD, HH, 2 * DD * HH);
  wfrag_kernel<<<(HH * DD + 255) / 256, 256, 0, stream>>>(nW2, nW2b, HH, DD, HH * DD);

  edge_kernel<<<NE / 32, 256, 0, stream>>>(nodes, edges, senders, receivers,
                                           eW1b, eb1, eg, ebt, eW2b, eb2,
                                           edges_out, agg);
  node_kernel<<<(NN + 31) / 32, 256, 0, stream>>>(nodes, agg,
                                                  nW1b, nb1, ng, nbt, nW2b, nb2,
                                                  nodes_out);
}
